// CommLayer_87067577024538
// MI455X (gfx1250) — matrix operand order repaired and checked by reading, not compiled, not run on hardware
//
#include <hip/hip_runtime.h>

// CDNA5 / gfx1250: wave32, WMMA 16x16x32 bf16
typedef __attribute__((ext_vector_type(16))) __bf16 v16bf;
typedef __attribute__((ext_vector_type(8)))  float  v8f;

#define AGENTS 8
#define DDIM   64
#define BPW    16                 // batches per workgroup
#define ROWS   (BPW * AGENTS)     // 128 agent-rows per workgroup
#define XS_STRIDE 36              // dwords per LDS row (32 data + 4 pad = 144B, 16B aligned)
#define BTILE_DW  (16 * 256)      // 16 B-fragment tiles x 256 dwords

__device__ __forceinline__ unsigned int pack2_bf16(float a, float b) {
  unsigned int ua = __float_as_uint(a);
  unsigned int ub = __float_as_uint(b);
  ua += 0x7FFFu + ((ua >> 16) & 1u);   // round-to-nearest-even
  ub += 0x7FFFu + ((ub >> 16) & 1u);
  return (ua >> 16) | (ub & 0xFFFF0000u);
}
__device__ __forceinline__ float bf16lo_to_f32(unsigned int u) { return __uint_as_float(u << 16); }
__device__ __forceinline__ float bf16hi_to_f32(unsigned int u) { return __uint_as_float(u & 0xFFFF0000u); }

// Branch-free tanh: sign(x) * (1 - e) / (1 + e), e = 2^(-2*log2(e)*|x|).
// e underflows to 0 for |x| > ~44 -> result +-1. Uses v_exp_f32 + v_rcp_f32.
__device__ __forceinline__ float fast_tanh(float v) {
  float ax = __builtin_fabsf(v);
  float e  = __builtin_amdgcn_exp2f(ax * -2.8853900817779268f);
  float r  = (1.0f - e) * __builtin_amdgcn_rcpf(1.0f + e);
  return __builtin_copysignf(r, v);
}

// Combined-weight element (logical WMMA B layout coordinates).
// K=128: k<64 -> (Wh - Wc/7), k>=64 -> Wc/7.  Tile t=kt*4+nt; lane l: group g=l/16,
// col n=l&15; VGPR v holds K = kt*32 + g*16 + 2v (+1 in high half).
__device__ __forceinline__ unsigned int btile_value(const float* wh, const float* wc,
                                                    int t, int l, int v) {
  int kt = t >> 2, nt = t & 3;
  int g  = l >> 4, n  = l & 15;
  int e  = nt * 16 + n;
  int k0 = kt * 32 + g * 16 + 2 * v;
  float w0, w1;
  if (k0 < 64) {
    w0 = wh[e * 64 + k0]     - wc[e * 64 + k0]     * (1.0f / 7.0f);
    w1 = wh[e * 64 + k0 + 1] - wc[e * 64 + k0 + 1] * (1.0f / 7.0f);
  } else {
    w0 = wc[e * 64 + (k0 - 64)] * (1.0f / 7.0f);
    w1 = wc[e * 64 + (k0 - 63)] * (1.0f / 7.0f);
  }
  return pack2_bf16(w0, w1);
}

// Physical (bank-swizzled) storage: within lane l's 8 dwords, the two 16B chunks
// are swapped when (l>>3)&1 == 1.  This makes each ds_load_b128 of the B fragment
// cover all 16 bank quads across lanes 0..15 (2-way overall = optimal).
__device__ __forceinline__ unsigned int btile_dword(const float* wh, const float* wc, int di) {
  int t   = di >> 8;
  int r   = di & 255;
  int l   = r >> 3;
  int p   = (r >> 2) & 1;            // physical chunk
  int vin = r & 3;
  int c   = p ^ ((l >> 3) & 1);      // logical chunk
  return btile_value(wh, wc, t, l, c * 4 + vin);
}

// One-shot prep: materialize packed, swizzled bf16 B fragments (16 KB) into workspace.
__global__ __launch_bounds__(256)
void prep_btile(const float* __restrict__ wh, const float* __restrict__ wc,
                unsigned int* __restrict__ bw) {
  int di = blockIdx.x * 256 + threadIdx.x;     // 0..4095
  bw[di] = btile_dword(wh, wc, di);
}

template <bool FROM_WS>
__global__ __launch_bounds__(256)
void comm_layer_wmma(const float* __restrict__ x,
                     const float* __restrict__ wh,
                     const float* __restrict__ wc,
                     const unsigned int* __restrict__ bws,
                     float* __restrict__ out) {
  __shared__ __align__(16) unsigned int xs[ROWS * XS_STRIDE];   // 18432 B
  __shared__ __align__(16) unsigned int ss[BPW * XS_STRIDE];    //  2304 B
  __shared__ __align__(16) unsigned int btile[BTILE_DW];        // 16384 B

  const int tid  = threadIdx.x;
  const int bat0 = blockIdx.x * BPW;

  // ---- stage X tile: 128 rows x 64 f32 -> bf16 in LDS (2048 float4 / 256 thr) ----
  const float4* xg = (const float4*)(x + (size_t)bat0 * (AGENTS * DDIM));
  #pragma unroll
  for (int i = 0; i < 8; ++i) {
    int fidx = tid + i * 256;        // 0..2047
    int row  = fidx >> 4;
    int c4   = fidx & 15;
    float4 v = xg[fidx];
    int di = row * XS_STRIDE + c4 * 2;
    xs[di]     = pack2_bf16(v.x, v.y);
    xs[di + 1] = pack2_bf16(v.z, v.w);
  }

  // ---- stage packed B fragments into LDS (layout already swizzled) ----
  if constexpr (FROM_WS) {
    const uint4* bg = (const uint4*)bws;
    #pragma unroll
    for (int j = 0; j < 4; ++j) {
      int qi = tid + j * 256;                  // 0..1023 uint4
      ((uint4*)btile)[qi] = bg[qi];
    }
  } else {
    #pragma unroll
    for (int j = 0; j < 16; ++j) {
      int di = tid + j * 256;                  // 0..4095
      btile[di] = btile_dword(wh, wc, di);
    }
  }

  __syncthreads();

  // ---- per-batch sums s_b = sum over 8 agents (16 batches x 32 dword-cols) ----
  #pragma unroll
  for (int j = 0; j < 2; ++j) {
    int tsk = tid + j * 256;         // 0..511
    int bi  = tsk >> 5;
    int cp  = tsk & 31;
    float f0 = 0.0f, f1 = 0.0f;
    #pragma unroll
    for (int r = 0; r < 8; ++r) {
      unsigned int u = xs[(bi * 8 + r) * XS_STRIDE + cp];
      f0 += bf16lo_to_f32(u);
      f1 += bf16hi_to_f32(u);
    }
    ss[bi * XS_STRIDE + cp] = pack2_bf16(f0, f1);
  }

  __syncthreads();

  // ---- WMMA: each wave computes a 16-row x 64-col output tile ----
  const int wv   = tid >> 5;
  const int lane = tid & 31;
  const int m = lane & 15, g = lane >> 4;
  const int sw = (lane >> 3) & 1;    // B-fragment chunk swizzle

  union AB { uint4 q[2]; v16bf v; };

  // A fragments for 4 k-tiles: 16-bit A layout = two contiguous 16B chunks/lane.
  AB a[4];
  #pragma unroll
  for (int kt = 0; kt < 4; ++kt) {
    const unsigned int* base;
    if (kt < 2) {                    // x part, K = 0..63
      int row = wv * 16 + m;
      base = &xs[row * XS_STRIDE + kt * 16 + g * 8];
    } else {                         // s part, K = 64..127, row = batch of m (broadcast)
      int brow = wv * 2 + (m >> 3);
      base = &ss[brow * XS_STRIDE + (kt - 2) * 16 + g * 8];
    }
    a[kt].q[0] = *(const uint4*)(base);
    a[kt].q[1] = *(const uint4*)(base + 4);
  }

  #pragma unroll
  for (int nt = 0; nt < 4; ++nt) {
    v8f acc = {};
    #pragma unroll
    for (int kt = 0; kt < 4; ++kt) {
      AB b;
      const unsigned int* bb = &btile[(kt * 4 + nt) * 256 + lane * 8];
      b.q[0] = *(const uint4*)(bb + sw * 4);        // logical chunk 0
      b.q[1] = *(const uint4*)(bb + (sw ^ 1) * 4);  // logical chunk 1
      acc = __builtin_amdgcn_wmma_f32_16x16x32_bf16(
          false, a[kt].v, false, b.v, (short)0, acc, false, false);
    }
    // D layout: VGPR i -> M = i + g*8, N = lane&15
    #pragma unroll
    for (int i = 0; i < 8; ++i) {
      int row_g = bat0 * AGENTS + wv * 16 + g * 8 + i;
      int col   = nt * 16 + m;
      out[(size_t)row_g * 64 + col] = fast_tanh(acc[i]);
    }
  }
}

extern "C" void kernel_launch(void* const* d_in, const int* in_sizes, int n_in,
                              void* d_out, int out_size, void* d_ws, size_t ws_size,
                              hipStream_t stream) {
  const float* x  = (const float*)d_in[0];
  const float* wh = (const float*)d_in[1];
  const float* wc = (const float*)d_in[2];
  float* out = (float*)d_out;
  int nbatch = in_sizes[0] / (AGENTS * DDIM);   // 65536
  dim3 grid(nbatch / BPW);                      // 4096 workgroups
  dim3 block(256);                              // 8 wave32

  if (d_ws != nullptr && ws_size >= BTILE_DW * sizeof(unsigned int)) {
    unsigned int* bw = (unsigned int*)d_ws;
    hipLaunchKernelGGL(prep_btile, dim3(16), block, 0, stream, wh, wc, bw);
    hipLaunchKernelGGL((comm_layer_wmma<true>), grid, block, 0, stream,
                       x, wh, wc, bw, out);
  } else {
    hipLaunchKernelGGL((comm_layer_wmma<false>), grid, block, 0, stream,
                       x, wh, wc, (const unsigned int*)nullptr, out);
  }
}
